// BiLSTM_CRF_28741921145427
// MI455X (gfx1250) — matrix operand order repaired and checked by reading, not compiled, not run on hardware
//
#include <hip/hip_runtime.h>
#include <hip/hip_bf16.h>
#include <hip/hip_fp16.h>

typedef __attribute__((ext_vector_type(16))) _Float16 v16h;
typedef __attribute__((ext_vector_type(8)))  _Float16 v8h;
typedef __attribute__((ext_vector_type(8)))  float    v8f;

#define T_LEN   8192
#define E_DIM   50
#define E_PAD   64
#define HH      128
#define G4      512      // 4*HH gates
#define KTAGS   23
#define KPAD    32
#define START_TAG 21
#define STOP_TAG  22
#define NEGV    (-10000.0f)

// ---------------------------------------------------------------------------
// Kernel 1: embedding gather -> f16 padded rows [T, 64]
// ---------------------------------------------------------------------------
__global__ void embed_gather(const int* __restrict__ sent,
                             const float* __restrict__ table,
                             _Float16* __restrict__ x16) {
    int t = blockIdx.x;
    int k = threadIdx.x;                 // 64 threads
    int tok = sent[t];
    float v = (k < E_DIM) ? table[(size_t)tok * E_DIM + k] : 0.0f;
    x16[t * E_PAD + k] = (_Float16)v;
}

// ---------------------------------------------------------------------------
// Kernel 2a: swizzle Wih (both dirs) into WMMA B-operand layout (f16).
// Layout: [dir][nTile(32)][kTile(2)][lane(32)][e(16)]
// B is K x N (K=E_PAD, N=gate). Contiguous k order: lane holds col n = lane&15;
// lanes 0-15: K = kTile*32 + e ; lanes 16-31: K = kTile*32 + 16 + e.
// ---------------------------------------------------------------------------
__global__ void swizzle_wih(const float* __restrict__ wf,
                            const float* __restrict__ wb,
                            _Float16* __restrict__ out) {
    int id   = blockIdx.x * blockDim.x + threadIdx.x;   // 65536 total
    int e    = id & 15;
    int lane = (id >> 4) & 31;
    int kt   = (id >> 9) & 1;
    int nt   = (id >> 10) & 31;
    int dir  = (id >> 15) & 1;
    int n = nt * 16 + (lane & 15);
    int k = kt * 32 + ((lane >> 4) << 4) + e;
    const float* w = dir ? wb : wf;
    float v = (k < E_DIM) ? w[n * E_DIM + k] : 0.0f;
    out[id] = (_Float16)v;
}

// ---------------------------------------------------------------------------
// Kernel 2b: swizzle Whh into WMMA A-operand layout (f16).
// Layout: [dir][mTile(32)][kTile(4)][lane(32)][e(16)]
// A is M x K (M=gate, K=HH). 16-bit A 16x32, contiguous k order:
// lanes 0-15: e0..15 -> K kbase+0..15
// lanes 16-31: shifted by +16.
// ---------------------------------------------------------------------------
__global__ void swizzle_whh(const float* __restrict__ wf,
                            const float* __restrict__ wb,
                            _Float16* __restrict__ out) {
    int id   = blockIdx.x * blockDim.x + threadIdx.x;   // 131072 total
    int e    = id & 15;
    int lane = (id >> 4) & 31;
    int kt   = (id >> 9) & 3;
    int mt   = (id >> 11) & 31;
    int dir  = (id >> 16) & 1;
    int m    = mt * 16 + (lane & 15);
    int off8 = (lane >> 4) << 3;
    int k = kt * 32 + (off8 << 1) + e;
    const float* w = dir ? wb : wf;
    out[id] = (_Float16)w[m * HH + k];
}

// ---------------------------------------------------------------------------
// Kernel 3: batched input projection preGate[dir][t][gate] =
//   Wih @ x_t + (bih + bhh), via v_wmma_f32_16x16x32_f16.
// grid (mTiles=512, nTiles=32, dir=2), one wave per block.
// ---------------------------------------------------------------------------
__global__ __launch_bounds__(32)
void input_gemm(const _Float16* __restrict__ x16,
                const _Float16* __restrict__ bswz,
                const float* __restrict__ bih_f, const float* __restrict__ bhh_f,
                const float* __restrict__ bih_b, const float* __restrict__ bhh_b,
                float* __restrict__ preGate) {
    int mt   = blockIdx.x;
    int nt   = blockIdx.y;
    int dir  = blockIdx.z;
    int lane = threadIdx.x;
    int trow = mt * 16 + (lane & 15);
    int off8 = (lane >> 4) << 3;

    v8f acc = {};
#pragma unroll
    for (int kt = 0; kt < 2; ++kt) {
        const _Float16* abase = x16 + trow * E_PAD + kt * 32;
        v8h lo = *(const v8h*)(abase + 2 * off8);
        v8h hi = *(const v8h*)(abase + 2 * off8 + 8);
        v16h a;
#pragma unroll
        for (int i = 0; i < 8; ++i) { a[i] = lo[i]; a[8 + i] = hi[i]; }
        const _Float16* bbase =
            bswz + ((((size_t)dir * 32 + nt) * 2 + kt) * 32 + lane) * 16;
        v16h b = *(const v16h*)bbase;
        acc = __builtin_amdgcn_wmma_f32_16x16x32_f16(
            false, a, false, b, (short)0, acc, false, false);
    }
    int col = nt * 16 + (lane & 15);
    const float* bi = dir ? bih_b : bih_f;
    const float* bh = dir ? bhh_b : bhh_f;
    float bias = bi[col] + bh[col];
    int rbase = mt * 16 + ((lane >> 4) << 3);   // D: vgpr r -> row r (+8 upper half)
    float* outp = preGate + (size_t)dir * T_LEN * G4;
#pragma unroll
    for (int r = 0; r < 8; ++r)
        outp[(size_t)(rbase + r) * G4 + col] = acc[r] + bias;
}

__device__ __forceinline__ float sigmoidf_(float x) {
    return 1.0f / (1.0f + __expf(-x));
}

// ---------------------------------------------------------------------------
// Kernel 4: LSTM recurrence. grid=2 (one block per direction), 512 threads.
// Whh resident in VGPRs as WMMA A-tiles; h broadcast via LDS (replicated into
// all 16 B columns); preGate loads software-pipelined one step ahead, with a
// global_prefetch 4 steps ahead to stage L2 lines into the WGP cache.
// ---------------------------------------------------------------------------
__global__ __launch_bounds__(512)
void lstm_rec(const _Float16* __restrict__ whhswz,
              const float* __restrict__ preGate,
              const float* __restrict__ h0, const float* __restrict__ c0,
              float* __restrict__ hcat) {
    int dir  = blockIdx.x;
    int tid  = threadIdx.x;
    int wave = tid >> 5;
    int lane = tid & 31;

    __shared__ __attribute__((aligned(64))) _Float16 hsh[HH];
    __shared__ __attribute__((aligned(64))) float    gsh[G4];

    // Stationary A tiles: wave w owns gate rows [w*32, w*32+31]
    v16h a[2][4];
#pragma unroll
    for (int i = 0; i < 2; ++i)
#pragma unroll
        for (int kt = 0; kt < 4; ++kt) {
            int mt = wave * 2 + i;
            a[i][kt] = *(const v16h*)(
                whhswz + ((((size_t)dir * 32 + mt) * 4 + kt) * 32 + lane) * 16);
        }

    float c_reg = 0.f, pgi = 0.f, pgf = 0.f, pgg = 0.f, pgo = 0.f;
    int j = tid;
    if (j < HH) {
        c_reg = c0[dir * HH + j];
        hsh[j] = (_Float16)h0[dir * HH + j];
    }
    int stride = dir ? -1 : 1;
    int tt = dir ? (T_LEN - 1) : 0;
    const float* pg = preGate + (size_t)dir * T_LEN * G4;
    if (j < HH) {
        const float* p = pg + (size_t)tt * G4;
        pgi = p[j]; pgf = p[HH + j]; pgg = p[2 * HH + j]; pgo = p[3 * HH + j];
    }

    for (int step = 0; step < T_LEN; ++step) {
        __syncthreads();                       // hsh from previous step visible
        v8f acc0 = {}, acc1 = {};
#pragma unroll
        for (int kt = 0; kt < 4; ++kt) {
            // B 32x16: lane holds column lane&15; lanes 0-15 K 0..15, 16-31 K 16..31
            v16h b = *(const v16h*)(hsh + kt * 32 + ((lane >> 4) << 4));
            acc0 = __builtin_amdgcn_wmma_f32_16x16x32_f16(
                false, a[0][kt], false, b, (short)0, acc0, false, false);
            acc1 = __builtin_amdgcn_wmma_f32_16x16x32_f16(
                false, a[1][kt], false, b, (short)0, acc1, false, false);
        }
        if ((lane & 15) == 0) {                // column 0 holds the matvec result
            int half = (lane >> 4) << 3;
            int m0 = (wave * 2 + 0) * 16 + half;
            int m1 = (wave * 2 + 1) * 16 + half;
#pragma unroll
            for (int r = 0; r < 8; ++r) { gsh[m0 + r] = acc0[r]; gsh[m1 + r] = acc1[r]; }
        }
        __syncthreads();
        if (j < HH) {
            float gi = sigmoidf_(pgi + gsh[j]);
            float gf = sigmoidf_(pgf + gsh[HH + j]);
            float gg = tanhf(pgg + gsh[2 * HH + j]);
            float go = sigmoidf_(pgo + gsh[3 * HH + j]);
            c_reg = gf * c_reg + gi * gg;
            float h = go * tanhf(c_reg);
            hsh[j] = (_Float16)h;
            hcat[(size_t)tt * (2 * HH) + dir * HH + j] = h;
            if (step + 1 < T_LEN) {            // pipeline next step's preGate
                const float* p = pg + (size_t)(tt + stride) * G4;
                pgi = p[j]; pgf = p[HH + j]; pgg = p[2 * HH + j]; pgo = p[3 * HH + j];
            }
            if (step + 4 < T_LEN) {            // stage L2 lines into WGP cache
                const float* pf = pg + (size_t)(tt + 4 * stride) * G4;
                __builtin_prefetch(pf + j, 0, 0);
                __builtin_prefetch(pf + 2 * HH + j, 0, 0);
            }
        }
        tt += stride;
    }
}

// ---------------------------------------------------------------------------
// Kernel 5: output projection feats[t][k] = W_out[k,:] . hcat[t,:] + b_out[k]
// 256 threads: 8 timesteps x 32 lanes (k padded to 32, pad = NEG).
// ---------------------------------------------------------------------------
__global__ __launch_bounds__(256)
void out_proj(const float* __restrict__ hcat,
              const float* __restrict__ Wout, const float* __restrict__ bout,
              float* __restrict__ feats) {
    __shared__ __attribute__((aligned(16))) float wsh[KTAGS * 2 * HH];
    for (int i = threadIdx.x; i < KTAGS * 2 * HH; i += 256) wsh[i] = Wout[i];
    __syncthreads();
    int tl = threadIdx.x >> 5;
    int k  = threadIdx.x & 31;
    int t  = blockIdx.x * 8 + tl;
    float acc = NEGV;
    if (k < KTAGS) {
        const float4* h4 = (const float4*)(hcat + (size_t)t * 2 * HH);
        const float4* w4 = (const float4*)(wsh + k * 2 * HH);
        acc = 0.0f;
#pragma unroll 4
        for (int jj = 0; jj < (2 * HH) / 4; ++jj) {
            float4 wv = w4[jj];
            float4 hv = h4[jj];
            acc += wv.x * hv.x + wv.y * hv.y + wv.z * hv.z + wv.w * hv.w;
        }
        acc += bout[k];
    }
    feats[(size_t)t * KPAD + k] = acc;
}

// ---------------------------------------------------------------------------
// Kernel 6: Viterbi. Single wave (barriers are free on single-wave workgroups).
// Forward pass with byte backpointers, then single-lane backtrace.
// ---------------------------------------------------------------------------
__global__ __launch_bounds__(32)
void viterbi(const float* __restrict__ feats,
             const float* __restrict__ trans,
             unsigned char* __restrict__ bp,
             float* __restrict__ out) {
    int k = threadIdx.x;
    __shared__ float fv[KPAD];
    __shared__ float red[KPAD];

    float tr[KTAGS];
#pragma unroll
    for (int p = 0; p < KTAGS; ++p) {
        // constrained: row START = NEG (no transition INTO START),
        // col STOP = NEG (no transition OUT OF STOP)
        float v = (k < KTAGS && k != START_TAG && p != STOP_TAG)
                      ? trans[k * KTAGS + p] : NEGV;
        tr[p] = v;
    }
    fv[k] = (k == START_TAG) ? 0.0f : NEGV;
    __syncthreads();

    for (int t = 0; t < T_LEN; ++t) {
        float best = -3.0e38f; int arg = 0;
#pragma unroll
        for (int p = 0; p < KTAGS; ++p) {
            float s = fv[p] + tr[p];
            if (s > best) { best = s; arg = p; }   // strict > keeps first argmax
        }
        float nv = best + feats[(size_t)t * KPAD + k];
        __syncthreads();
        if (k < KTAGS) fv[k] = nv;
        bp[(size_t)t * KPAD + k] = (unsigned char)arg;
        __syncthreads();
    }

    float term = -3.0e38f;
    if (k < KTAGS)
        term = fv[k] + ((k == STOP_TAG) ? NEGV : trans[STOP_TAG * KTAGS + k]);
    red[k] = term;
    __syncthreads();
    if (k == 0) {
        float best = red[0]; int bi = 0;
        for (int p = 1; p < KTAGS; ++p)
            if (red[p] > best) { best = red[p]; bi = p; }
        out[0] = best;
        int tag = bi;
        out[T_LEN] = (float)tag;                 // out[1 + (T-1)]
        for (int t = T_LEN - 1; t >= 1; --t) {
            tag = bp[(size_t)t * KPAD + tag];
            out[t] = (float)tag;                 // out[1 + (t-1)]
        }
    }
}

// ---------------------------------------------------------------------------
extern "C" void kernel_launch(void* const* d_in, const int* in_sizes, int n_in,
                              void* d_out, int out_size, void* d_ws, size_t ws_size,
                              hipStream_t stream) {
    (void)in_sizes; (void)n_in; (void)out_size; (void)ws_size;
    const int*   sent  = (const int*)d_in[0];
    const float* table = (const float*)d_in[1];
    const float* Wih_f = (const float*)d_in[2];
    const float* Whh_f = (const float*)d_in[3];
    const float* bih_f = (const float*)d_in[4];
    const float* bhh_f = (const float*)d_in[5];
    const float* Wih_b = (const float*)d_in[6];
    const float* Whh_b = (const float*)d_in[7];
    const float* bih_b = (const float*)d_in[8];
    const float* bhh_b = (const float*)d_in[9];
    const float* W_out = (const float*)d_in[10];
    const float* b_out = (const float*)d_in[11];
    const float* trans = (const float*)d_in[12];
    const float* h0    = (const float*)d_in[13];
    const float* c0    = (const float*)d_in[14];

    char* ws = (char*)d_ws;
    _Float16* x16     = (_Float16*)(ws + 0);          //  1,048,576 B
    _Float16* bihswz  = (_Float16*)(ws + 1048576);    //    131,072 B
    _Float16* whhswz  = (_Float16*)(ws + 1179648);    //    524,288 B
    float*    preGate = (float*)   (ws + 1703936);    // 33,554,432 B
    float*    hcat    = (float*)   (ws + 35258368);   //  8,388,608 B
    float*    feats   = (float*)   (ws + 43646976);   //  1,048,576 B
    unsigned char* bp = (unsigned char*)(ws + 44695552); // 262,144 B

    embed_gather<<<T_LEN, 64, 0, stream>>>(sent, table, x16);
    swizzle_wih<<<256, 256, 0, stream>>>(Wih_f, Wih_b, bihswz);
    swizzle_whh<<<512, 256, 0, stream>>>(Whh_f, Whh_b, whhswz);
    dim3 g3(512, 32, 2);
    input_gemm<<<g3, 32, 0, stream>>>(x16, bihswz, bih_f, bhh_f, bih_b, bhh_b,
                                      preGate);
    lstm_rec<<<2, 512, 0, stream>>>(whhswz, preGate, h0, c0, hcat);
    out_proj<<<T_LEN / 8, 256, 0, stream>>>(hcat, W_out, b_out, feats);
    viterbi<<<1, 32, 0, stream>>>(feats, trans, bp, (float*)d_out);
}
